// LogicNetwork_15702400434248
// MI455X (gfx1250) — matrix operand order repaired and checked by reading, not compiled, not run on hardware
//
#include <hip/hip_runtime.h>
#include <math.h>

// ---------------------------------------------------------------------------
// LogicNetwork soft-OR product:
//   out[b,o] = prod_i ( 1 - (1 - atoms[b,i]) * sigmoid(weights[o,i]) )
// via log/power-series -> one f16 WMMA GEMM with exp(-x) epilogue:
//   out[b,o] = exp( - sum_k (1/k) sum_i (1-a)^k * fw^k ),  k = 1..KPOW
// Staged pre-swizzled f16 operands live in d_ws; GEMM stages per-k-tile
// fragments block-wide into LDS with a 2-deep async-to-LDS pipeline
// (triple-buffered, ASYNCcnt-throttled).
// ---------------------------------------------------------------------------

typedef __attribute__((ext_vector_type(16))) _Float16 v16h;
typedef __attribute__((ext_vector_type(8)))  float    v8f;
typedef __attribute__((ext_vector_type(4)))  int      v4i;

#define B_DIM   512
#define IN_DIM  1024
#define OUT_DIM 1024
#define KPOW    8
#define KDIM    (KPOW * IN_DIM)   // 8192
#define NKT     (KDIM / 32)       // 256 k-tiles of 32
#define FRAG_BYTES 1024           // one 16x32 (or 32x16) f16 fragment
#define BLK_FRAGS  8              // 4 A-frags + 4 B-frags per k-step

#if defined(__AMDGCN__) && __has_builtin(__builtin_amdgcn_global_load_async_to_lds_b128)
#define HAVE_ASYNC_LDS 1
#else
#define HAVE_ASYNC_LDS 0
#endif

#if HAVE_ASYNC_LDS
__device__ __forceinline__ void async_copy16(const void* g, void* l) {
  __builtin_amdgcn_global_load_async_to_lds_b128(
      (__attribute__((address_space(1))) v4i*)(g),
      (__attribute__((address_space(3))) v4i*)(l),
      /*offset=*/0, /*cpol=*/0);
}
__device__ __forceinline__ void wait_async0() {
#if __has_builtin(__builtin_amdgcn_s_wait_asynccnt)
  __builtin_amdgcn_s_wait_asynccnt(0);
#else
  asm volatile("s_wait_asynccnt 0" ::: "memory");
#endif
}
__device__ __forceinline__ void wait_async4() {
#if __has_builtin(__builtin_amdgcn_s_wait_asynccnt)
  __builtin_amdgcn_s_wait_asynccnt(4);
#else
  asm volatile("s_wait_asynccnt 4" ::: "memory");
#endif
}
#endif

// ---------------------------------------------------------------------------
// Prep A: astg tile (mt, kt) = 16(M) x 32(K) fragment, contiguous-k A lane
// layout: lane l (m = l&15, kb = (l>>4)*16): halves[0..7] = K kb..kb+7,
// halves[8..15] = K kb+8..kb+15.  A'[b,(p-1)*IN+i] = (1-atoms[b,i])^p / p.
// ---------------------------------------------------------------------------
__global__ __launch_bounds__(256) void prep_a_kernel(
    const float* __restrict__ atoms, _Float16* __restrict__ astg) {
  const int t    = blockIdx.x * blockDim.x + threadIdx.x;
  const int lane = t & 31;
  const int tile = t >> 5;
  const int mt   = tile / NKT;
  const int kt   = tile - mt * NKT;
  const int m    = mt * 16 + (lane & 15);
  const int kg   = kt * 32;
  const int p    = kg / IN_DIM + 1;
  const float invp  = 1.0f / (float)p;
  const int   ibase = (kg & (IN_DIM - 1)) + ((lane >> 4) << 4);
  const float* arow = atoms + (size_t)m * IN_DIM;

  v16h frag;
#pragma unroll
  for (int h = 0; h < 8; ++h) {
    float oma = 1.0f - arow[ibase + h];
    float v = oma;
    for (int j = 1; j < p; ++j) v *= oma;
    frag[h] = (_Float16)(v * invp);
  }
#pragma unroll
  for (int h = 0; h < 8; ++h) {
    float oma = 1.0f - arow[ibase + 8 + h];
    float v = oma;
    for (int j = 1; j < p; ++j) v *= oma;
    frag[8 + h] = (_Float16)(v * invp);
  }
  ((v16h*)astg)[(size_t)tile * 32 + lane] = frag;
}

// ---------------------------------------------------------------------------
// Prep W: wstg tile (nt, kt) = 32(K) x 16(N) fragment, contiguous-k B lane layout (same k order as A):
// lane l (n = l&15, kb = (l>>4)*16): halves[0..15] = K kb..kb+15.
// W'[o,(p-1)*IN+i] = sigmoid(weights[o,i])^p.
// ---------------------------------------------------------------------------
__global__ __launch_bounds__(256) void prep_w_kernel(
    const float* __restrict__ wts, _Float16* __restrict__ wstg) {
  const int t    = blockIdx.x * blockDim.x + threadIdx.x;
  const int lane = t & 31;
  const int tile = t >> 5;
  const int nt   = tile / NKT;
  const int kt   = tile - nt * NKT;
  const int n    = nt * 16 + (lane & 15);
  const int kg   = kt * 32;
  const int p    = kg / IN_DIM + 1;
  const int ibase = (kg & (IN_DIM - 1)) + ((lane >> 4) << 4);
  const float* wrow = wts + (size_t)n * IN_DIM;

  v16h frag;
#pragma unroll
  for (int h = 0; h < 16; ++h) {
    float w  = wrow[ibase + h];
    float fw = 1.0f / (1.0f + expf(-w));
    float v = fw;
    for (int j = 1; j < p; ++j) v *= fw;
    frag[h] = (_Float16)v;
  }
  ((v16h*)wstg)[(size_t)tile * 32 + lane] = frag;
}

// ---------------------------------------------------------------------------
// GEMM + exp(-x) epilogue.
// Block: 128 threads = 4 waves, block tile 64(M) x 64(N); wave (wm,wn) owns a
// 32x32 tile (2x2 fragments -> 4 WMMAs per k-step of 32).
// Per k-step the block stages its 8 unique fragments (8 KB) once into LDS.
// Async pipeline: 2 stages in flight, triple-buffered; s_wait_asynccnt 4
// retires exactly the oldest stage (async loads complete in order).
// Grid: (OUT/64, B/64) = (16, 8) = 128 blocks = 512 waves.
// ---------------------------------------------------------------------------
__global__ __launch_bounds__(128) void logic_gemm_kernel(
    const _Float16* __restrict__ astg, const _Float16* __restrict__ wstg,
    float* __restrict__ out) {
  __shared__ _Float16 smem[3][BLK_FRAGS * 512];   // 3 x 8 KB

  const int t    = threadIdx.x;
  const int lane = t & 31;
  const int w    = t >> 5;
  const int wm   = w >> 1;                 // 0..1
  const int wn   = w & 1;                  // 0..1
  const int mtB  = blockIdx.y * 4;         // base 16-row fragment index
  const int ntB  = blockIdx.x * 4;         // base 16-col fragment index

  // ---- per-thread copy chunks: 8 KB / 128 threads = 4 x 16 B -------------
  // chunk c = j*128 + t; frag f = c>>6 (0..3 -> A, 4..7 -> B);
  // global src = frag tile (row*NKT + kt)*1024 + (c&63)*16.
  const char* gsrc[4];
  int         loff[4];
#pragma unroll
  for (int j = 0; j < 4; ++j) {
    int c = j * 128 + t;
    int f = c >> 6;
    int intra = (c & 63) << 4;
    size_t row = (f < 4) ? (size_t)(mtB + f) : (size_t)(ntB + (f - 4));
    const char* bse = (f < 4) ? (const char*)astg : (const char*)wstg;
    gsrc[j] = bse + row * ((size_t)NKT * FRAG_BYTES) + intra;
    loff[j] = c << 4;
  }

  v8f c00 = {}, c01 = {}, c10 = {}, c11 = {};

  const int fa0 = (wm * 2 + 0) * 32;       // v16h index of wave's fragments
  const int fa1 = (wm * 2 + 1) * 32;
  const int fb0 = (4 + wn * 2 + 0) * 32;
  const int fb1 = (4 + wn * 2 + 1) * 32;

#if HAVE_ASYNC_LDS
  char* pc = (char*)&smem[0][0];           // compute buffer (stage kt)
  char* pn = (char*)&smem[1][0];           // next (stage kt+1, in flight)
  char* pp = (char*)&smem[2][0];           // prefetch target (stage kt+2)
#pragma unroll
  for (int j = 0; j < 4; ++j) async_copy16(gsrc[j], pc + loff[j]);
#pragma unroll
  for (int j = 0; j < 4; ++j) async_copy16(gsrc[j] + FRAG_BYTES, pn + loff[j]);

  for (int kt = 0; kt < NKT; ++kt) {
    // Retire stage kt: 4 newer copy instructions (stage kt+1) may remain.
    if (kt + 1 < NKT) wait_async4();
    else              wait_async0();
    __syncthreads();   // all waves' stage-kt copies done; reads of pp from
                       // iteration kt-1 are complete -> safe to overwrite pp
    if (kt + 2 < NKT) {
      const size_t gofs = (size_t)(kt + 2) * FRAG_BYTES;
#pragma unroll
      for (int j = 0; j < 4; ++j) async_copy16(gsrc[j] + gofs, pp + loff[j]);
    }
    const v16h* sb = (const v16h*)pc;
    v16h a0 = sb[fa0 + lane];
    v16h a1 = sb[fa1 + lane];
    v16h b0 = sb[fb0 + lane];
    v16h b1 = sb[fb1 + lane];
    c00 = __builtin_amdgcn_wmma_f32_16x16x32_f16(false, a0, false, b0,
                                                 (short)0, c00, false, false);
    c01 = __builtin_amdgcn_wmma_f32_16x16x32_f16(false, a0, false, b1,
                                                 (short)0, c01, false, false);
    c10 = __builtin_amdgcn_wmma_f32_16x16x32_f16(false, a1, false, b0,
                                                 (short)0, c10, false, false);
    c11 = __builtin_amdgcn_wmma_f32_16x16x32_f16(false, a1, false, b1,
                                                 (short)0, c11, false, false);
    char* tmp = pc; pc = pn; pn = pp; pp = tmp;   // rotate buffers
  }
#else
  // Fallback pipeline: global->VGPR->ds_store, double-buffered.
  int4 r0, r1, r2, r3;
  r0 = *(const int4*)(gsrc[0]); r1 = *(const int4*)(gsrc[1]);
  r2 = *(const int4*)(gsrc[2]); r3 = *(const int4*)(gsrc[3]);
  {
    char* s0 = (char*)&smem[0][0];
    *(int4*)(s0 + loff[0]) = r0; *(int4*)(s0 + loff[1]) = r1;
    *(int4*)(s0 + loff[2]) = r2; *(int4*)(s0 + loff[3]) = r3;
  }
#pragma unroll 2
  for (int kt = 0; kt < NKT; ++kt) {
    __syncthreads();
    const bool pre = (kt + 1 < NKT);
    if (pre) {
      const size_t gofs = (size_t)(kt + 1) * FRAG_BYTES;
      r0 = *(const int4*)(gsrc[0] + gofs); r1 = *(const int4*)(gsrc[1] + gofs);
      r2 = *(const int4*)(gsrc[2] + gofs); r3 = *(const int4*)(gsrc[3] + gofs);
    }
    const v16h* sb = (const v16h*)&smem[kt & 1][0];
    v16h a0 = sb[fa0 + lane];
    v16h a1 = sb[fa1 + lane];
    v16h b0 = sb[fb0 + lane];
    v16h b1 = sb[fb1 + lane];
    c00 = __builtin_amdgcn_wmma_f32_16x16x32_f16(false, a0, false, b0,
                                                 (short)0, c00, false, false);
    c01 = __builtin_amdgcn_wmma_f32_16x16x32_f16(false, a0, false, b1,
                                                 (short)0, c01, false, false);
    c10 = __builtin_amdgcn_wmma_f32_16x16x32_f16(false, a1, false, b0,
                                                 (short)0, c10, false, false);
    c11 = __builtin_amdgcn_wmma_f32_16x16x32_f16(false, a1, false, b1,
                                                 (short)0, c11, false, false);
    if (pre) {
      char* sn = (char*)&smem[(kt + 1) & 1][0];
      *(int4*)(sn + loff[0]) = r0; *(int4*)(sn + loff[1]) = r1;
      *(int4*)(sn + loff[2]) = r2; *(int4*)(sn + loff[3]) = r3;
    }
  }
#endif

  // C/D layout: lane = {n = lane&15, m = 8*(lane>>4) + v}, v = VGPR 0..7.
  const int mt0 = mtB + wm * 2;
  const int nt0 = ntB + wn * 2;
  const int mr  = (lane >> 4) << 3;
  const int nc  = lane & 15;
  float* o00 = out + (size_t)(mt0 * 16 + mr) * OUT_DIM + nt0 * 16 + nc;
  float* o10 = o00 + (size_t)16 * OUT_DIM;
#pragma unroll
  for (int v = 0; v < 8; ++v) {
    o00[(size_t)v * OUT_DIM +  0] = expf(-c00[v]);
    o00[(size_t)v * OUT_DIM + 16] = expf(-c01[v]);
    o10[(size_t)v * OUT_DIM +  0] = expf(-c10[v]);
    o10[(size_t)v * OUT_DIM + 16] = expf(-c11[v]);
  }
}

// ---------------------------------------------------------------------------
// Exact VALU fallback (only if the workspace is too small for staging).
// ---------------------------------------------------------------------------
__global__ __launch_bounds__(256) void logic_fallback_kernel(
    const float* __restrict__ atoms, const float* __restrict__ wts,
    float* __restrict__ out) {
  const int o = blockIdx.x * blockDim.x + threadIdx.x;
  const int b = blockIdx.y;
  if (o >= OUT_DIM) return;
  const float* a  = atoms + (size_t)b * IN_DIM;
  const float* wr = wts + (size_t)o * IN_DIM;
  float prod = 1.0f;
  for (int i = 0; i < IN_DIM; ++i) {
    float fw = 1.0f / (1.0f + expf(-wr[i]));
    float t  = fmaf(-(1.0f - a[i]), fw, 1.0f);
    prod *= t;
  }
  out[(size_t)b * OUT_DIM + o] = prod;
}

extern "C" void kernel_launch(void* const* d_in, const int* in_sizes, int n_in,
                              void* d_out, int out_size, void* d_ws, size_t ws_size,
                              hipStream_t stream) {
  const float* atoms = (const float*)d_in[0];   // (B, IN)  f32
  const float* wts   = (const float*)d_in[1];   // (OUT, IN) f32
  float* out = (float*)d_out;                   // (B, OUT) f32

  const size_t a_halves = (size_t)B_DIM * KDIM;     // 8 MB
  const size_t w_halves = (size_t)OUT_DIM * KDIM;   // 16 MB
  const size_t need = (a_halves + w_halves) * sizeof(_Float16);

  if (ws_size >= need) {
    _Float16* astg = (_Float16*)d_ws;
    _Float16* wstg = astg + a_halves;

    const int aTiles = (B_DIM / 16) * NKT;          // 8192
    const int wTiles = (OUT_DIM / 16) * NKT;        // 16384
    prep_a_kernel<<<aTiles * 32 / 256, 256, 0, stream>>>(atoms, astg);
    prep_w_kernel<<<wTiles * 32 / 256, 256, 0, stream>>>(wts, wstg);

    dim3 grid(OUT_DIM / 64, B_DIM / 64);            // (16, 8) = 128 blocks
    logic_gemm_kernel<<<grid, 128, 0, stream>>>(astg, wstg, out);
  } else {
    dim3 grid(OUT_DIM / 256, B_DIM);
    logic_fallback_kernel<<<grid, 256, 0, stream>>>(atoms, wts, out);
  }
}
